// NodeScalarFeatInit_13176959664201
// MI455X (gfx1250) — matrix operand order repaired and checked by reading, not compiled, not run on hardware
//
#include <hip/hip_runtime.h>
#include <hip/hip_bf16.h>

// Problem constants (match reference)
#define BB 4
#define NN 512
#define DD 128
#define HH 64

typedef __attribute__((ext_vector_type(16))) _Float16 v16h;
typedef __attribute__((ext_vector_type(8)))  float    v8f;

union Frag8 {
    uint4    q[2];
    unsigned u[8];
    v16h     h;
};

__device__ __forceinline__ unsigned pkh(float a, float b) {
    union { unsigned u; _Float16 h[2]; } t;
    t.h[0] = (_Float16)a; t.h[1] = (_Float16)b;
    return t.u;
}
__device__ __forceinline__ float silu_f(float x) {
    return x / (1.0f + __expf(-x));
}
__device__ __forceinline__ uint4 lds_q(const unsigned* p) {
    return *reinterpret_cast<const uint4*>(p);
}

// ---------------------------------------------------------------------------
// Kernel 1: materialize neighbor embeddings nembB[b,j,d] = neigh_tab[atoms[b,j],d]
// ---------------------------------------------------------------------------
__global__ void nsf_prep_nemb(const int* __restrict__ atoms,
                              const float* __restrict__ neigh_tab,
                              float* __restrict__ nembB) {
    int bj = blockIdx.x;                 // 0 .. B*N-1
    int d  = threadIdx.x;                // 0 .. 127
    int a  = atoms[bj];
    nembB[bj * DD + d] = neigh_tab[a * DD + d];
}

// ---------------------------------------------------------------------------
// Kernel 2: fused pairwise radial MLP + masked einsum reduction over j.
// One block per (b,i). 256 threads = 8 waves; wave w handles j in [64w,64w+64).
// LDS layouts:
//   hbuf  : per-wave 16x64 f16 activations, row stride 72 halves (36 words)
//   pk2t  : rW2 pair-packed fragment-major: pk2t[n*36 + kp], n<64,  kp<32
//   pk3t  : rW3 pair-packed fragment-major: pk3t[n*36 + kp], n<128, kp<32
// B-fragment for (s, nt): lane reads 2x4 contiguous words at
//   (nt*16 + lane%16)*36 + 16*s + (lane>=16 ? 4 : 0)  (+8 for upper quad)
// A-fragment for s: 2x4 contiguous words at
//   row*36 + 16*s + (lane>=16 ? 4 : 0)  (+8 for upper quad)
// ---------------------------------------------------------------------------
__global__ void __launch_bounds__(256)
nsf_pair_kernel(const float* __restrict__ rel_dist,          // [B,N,N]
                const unsigned char* __restrict__ adj_mat,   // [B,N,N] bool
                const unsigned char* __restrict__ mask,      // [B,N]   bool
                const float* __restrict__ softmask,          // [B,N,N]
                const float* __restrict__ rW1,               // [64]
                const float* __restrict__ rb1,
                const float* __restrict__ rg1,
                const float* __restrict__ rW2,               // [64,64]
                const float* __restrict__ rb2,
                const float* __restrict__ rg2,
                const float* __restrict__ rW3,               // [64,128]
                const float* __restrict__ rb3,               // [128]
                const float* __restrict__ nembB,             // [B*N,128]
                float* __restrict__ nf_out)                  // [B*N,128]
{
    __shared__ __align__(16) _Float16 hbuf[8][16 * 72];
    __shared__ __align__(16) unsigned pk2t[64 * 36];
    __shared__ __align__(16) unsigned pk3t[128 * 36];
    __shared__ float wls[NN];
    __shared__ float sacc[DD];
    __shared__ float w1s[HH], b1s[HH], g1s[HH], b2s[HH], g2s[HH], b3s[DD];

    const int tid  = threadIdx.x;
    const int lane = tid & 31;
    const int wave = tid >> 5;
    const int bi = blockIdx.x;
    const int b = bi >> 9;               // / N
    const int i = bi & (NN - 1);

    const float* relrow = rel_dist + (size_t)bi * NN;
    __builtin_prefetch(relrow + tid * 2, 0, 0);   // global_prefetch_b8

    // ---- stage small params + packed weights + pair weights -------------
    if (tid < HH) {
        w1s[tid] = rW1[tid]; b1s[tid] = rb1[tid]; g1s[tid] = rg1[tid];
        b2s[tid] = rb2[tid]; g2s[tid] = rg2[tid];
    }
    if (tid < DD) { b3s[tid] = rb3[tid]; sacc[tid] = 0.0f; }
    for (int idx = tid; idx < 64 * 32; idx += 256) {
        int n = idx >> 5, kp = idx & 31;
        pk2t[n * 36 + kp] = pkh(rW2[(2 * kp) * 64 + n], rW2[(2 * kp + 1) * 64 + n]);
    }
    for (int idx = tid; idx < 128 * 32; idx += 256) {
        int n = idx >> 5, kp = idx & 31;
        pk3t[n * 36 + kp] = pkh(rW3[(2 * kp) * 128 + n], rW3[(2 * kp + 1) * 128 + n]);
    }
    for (int j = tid; j < NN; j += 256) {
        unsigned char a = adj_mat[(size_t)bi * NN + j];
        unsigned char m = mask[b * NN + j];
        float w = (a && m && (j != i)) ? softmask[(size_t)bi * NN + j] : 0.0f;
        wls[j] = w;
    }
    __syncthreads();

    _Float16* hb = hbuf[wave];
    unsigned* hb32 = reinterpret_cast<unsigned*>(hb);
    const int rloc16 = lane & 15;
    const int hiHalf = (lane >= 16);
    const float inv64 = 1.0f / 64.0f;
    const float eps = 1e-5f;

    float nfp[8];
#pragma unroll
    for (int nt = 0; nt < 8; ++nt) nfp[nt] = 0.0f;

    const int jwb = wave * 64;

    for (int mt = 0; mt < 4; ++mt) {
        const int jb = jwb + mt * 16;

        // ---- h1 = LN(silu(x*W1 + b1)): lane handles half a row ----------
        {
            const int koff = hiHalf ? 32 : 0;
            float x = relrow[jb + rloc16];
            float vals[32];
            float sm = 0.0f, ss = 0.0f;
#pragma unroll
            for (int t = 0; t < 32; ++t) {
                int k = koff + t;
                float v = silu_f(fmaf(x, w1s[k], b1s[k]));
                vals[t] = v; sm += v; ss += v * v;
            }
            sm += __shfl_xor(sm, 16, 32);
            ss += __shfl_xor(ss, 16, 32);
            float mean = sm * inv64;
            float var = ss * inv64 - mean * mean;
            float inv = rsqrtf(var + eps);
            const int wbase = rloc16 * 36 + (hiHalf ? 16 : 0);
#pragma unroll
            for (int t2 = 0; t2 < 16; ++t2) {
                int k = koff + 2 * t2;
                float a = (vals[2 * t2] - mean) * inv * g1s[k];
                float c = (vals[2 * t2 + 1] - mean) * inv * g1s[k + 1];
                hb32[wbase + t2] = pkh(a, c);
            }
        }

        // ---- GEMM1: [16x64] @ [64x64] -> acc1[4] tiles ------------------
        v8f acc1[4] = {};
#pragma unroll
        for (int s = 0; s < 2; ++s) {
            Frag8 af;
            const int abase = rloc16 * 36 + 16 * s + (hiHalf ? 4 : 0);
            af.q[0] = lds_q(&hb32[abase]);
            af.q[1] = lds_q(&hb32[abase + 8]);
#pragma unroll
            for (int nt = 0; nt < 4; ++nt) {
                Frag8 bf;
                const int bbase = (nt * 16 + rloc16) * 36 + 16 * s + (hiHalf ? 4 : 0);
                bf.q[0] = lds_q(&pk2t[bbase]);
                bf.q[1] = lds_q(&pk2t[bbase + 8]);
                acc1[nt] = __builtin_amdgcn_wmma_f32_16x16x32_f16(
                    false, af.h, false, bf.h, (short)0, acc1[nt], false, false);
            }
        }

        // ---- h2 = LN(silu(acc1 + rb2)) over 64 cols ---------------------
        {
            float sv[4][8];
            float rs[8], rq[8];
#pragma unroll
            for (int r = 0; r < 8; ++r) { rs[r] = 0.0f; rq[r] = 0.0f; }
#pragma unroll
            for (int nt = 0; nt < 4; ++nt) {
                int col = nt * 16 + rloc16;
                float bb = b2s[col];
#pragma unroll
                for (int r = 0; r < 8; ++r) {
                    float t = silu_f(acc1[nt][r] + bb);
                    sv[nt][r] = t; rs[r] += t; rq[r] += t * t;
                }
            }
#pragma unroll
            for (int r = 0; r < 8; ++r) {
                rs[r] += __shfl_xor(rs[r], 1, 32);
                rs[r] += __shfl_xor(rs[r], 2, 32);
                rs[r] += __shfl_xor(rs[r], 4, 32);
                rs[r] += __shfl_xor(rs[r], 8, 32);
                rq[r] += __shfl_xor(rq[r], 1, 32);
                rq[r] += __shfl_xor(rq[r], 2, 32);
                rq[r] += __shfl_xor(rq[r], 4, 32);
                rq[r] += __shfl_xor(rq[r], 8, 32);
            }
#pragma unroll
            for (int r = 0; r < 8; ++r) {
                float mean = rs[r] * inv64;
                float var = rq[r] * inv64 - mean * mean;
                float inv = rsqrtf(var + eps);
                int row2 = r + (hiHalf ? 8 : 0);
#pragma unroll
                for (int nt = 0; nt < 4; ++nt) {
                    int col = nt * 16 + rloc16;
                    hb[row2 * 72 + col] =
                        (_Float16)((sv[nt][r] - mean) * inv * g2s[col]);
                }
            }
        }

        // ---- GEMM2: [16x64] @ [64x128] -> feats tiles -------------------
        v8f acc2[8] = {};
#pragma unroll
        for (int s = 0; s < 2; ++s) {
            Frag8 af;
            const int abase = rloc16 * 36 + 16 * s + (hiHalf ? 4 : 0);
            af.q[0] = lds_q(&hb32[abase]);
            af.q[1] = lds_q(&hb32[abase + 8]);
#pragma unroll
            for (int nt = 0; nt < 8; ++nt) {
                Frag8 bf;
                const int bbase = (nt * 16 + rloc16) * 36 + 16 * s + (hiHalf ? 4 : 0);
                bf.q[0] = lds_q(&pk3t[bbase]);
                bf.q[1] = lds_q(&pk3t[bbase + 8]);
                acc2[nt] = __builtin_amdgcn_wmma_f32_16x16x32_f16(
                    false, af.h, false, bf.h, (short)0, acc2[nt], false, false);
            }
        }

        // ---- epilogue: fused (+rb3) * adj*mask*softmask * nemb, sum_j ---
#pragma unroll
        for (int nt = 0; nt < 8; ++nt) {
            int d = nt * 16 + rloc16;
            float b3 = b3s[d];
#pragma unroll
            for (int r = 0; r < 8; ++r) {
                int jloc = jb + r + (hiHalf ? 8 : 0);
                float w = wls[jloc];
                float ne = nembB[(size_t)(b * NN + jloc) * DD + d];
                nfp[nt] += (acc2[nt][r] + b3) * w * ne;
            }
        }
    }

    // ---- cross-wave reduce into sacc, write neighbor_feats --------------
#pragma unroll
    for (int nt = 0; nt < 8; ++nt) {
        float t = nfp[nt] + __shfl_xor(nfp[nt], 16, 32);
        if (lane < 16) atomicAdd(&sacc[nt * 16 + lane], t);
    }
    __syncthreads();
    if (tid < DD) nf_out[(size_t)bi * DD + tid] = sacc[tid];
}

// ---------------------------------------------------------------------------
// Kernel 3: node MLP. x = concat(embeds, nf) [2048,256];
// out = silu(LN(x@nW1+nb1))@nW2 + nb2. 32 blocks x 128 threads (4 waves).
// pkt: per-K-slice B staging, fragment-major pair-packed: pkt[n*20 + kp],
//      kp<16 (one 32-K slice), 16B-aligned fragment bases, conflict-free.
// ---------------------------------------------------------------------------
__global__ void __launch_bounds__(128)
nsf_node_kernel(const int* __restrict__ atoms,
                const float* __restrict__ atom_tab,    // [A,128]
                const float* __restrict__ nf,          // [B*N,128]
                const float* __restrict__ nW1,         // [256,128]
                const float* __restrict__ nb1,
                const float* __restrict__ ng,
                const float* __restrict__ nW2,         // [128,128]
                const float* __restrict__ nb2,
                float* __restrict__ out)               // [B*N,128]
{
    __shared__ __align__(16) _Float16 Abuf[4][16 * 264];  // row stride 132 words
    __shared__ __align__(16) unsigned pkt[128 * 20];
    __shared__ float nb1s[DD], ngs[DD], nb2s[DD];

    const int tid = threadIdx.x;
    const int lane = tid & 31;
    const int wave = tid >> 5;
    const int rowblock = blockIdx.x * 64;
    const int rloc16 = lane & 15;
    const int hiHalf = (lane >= 16);

    if (tid < DD) { nb1s[tid] = nb1[tid]; ngs[tid] = ng[tid]; nb2s[tid] = nb2[tid]; }

    // build A = concat(atom_tab[atoms[row]], nf[row]) in f16, packed u32 stores
    unsigned* Ab32all = reinterpret_cast<unsigned*>(&Abuf[0][0]);
    for (int e2 = tid; e2 < 64 * 128; e2 += 128) {
        int rloc = e2 >> 7, kp = e2 & 127;
        int row = rowblock + rloc;
        int k0 = 2 * kp;
        float v0, v1;
        if (k0 < 128) {
            const float* src = atom_tab + atoms[row] * DD;
            v0 = src[k0]; v1 = src[k0 + 1];
        } else {
            const float* src = nf + (size_t)row * DD - 128;
            v0 = src[k0]; v1 = src[k0 + 1];
        }
        Ab32all[(rloc >> 4) * (16 * 132) + (rloc & 15) * 132 + kp] = pkh(v0, v1);
    }

    unsigned* Ab32 = reinterpret_cast<unsigned*>(Abuf[wave]);

    // ---- GEMM1: K=256 in 8 k-slices ------------------------------------
    v8f acc[8] = {};
    for (int s = 0; s < 8; ++s) {
        __syncthreads();
        for (int idx = tid; idx < 128 * 16; idx += 128) {
            int n = idx >> 4, kpl = idx & 15;
            int kg = 32 * s + 2 * kpl;
            pkt[n * 20 + kpl] = pkh(nW1[kg * DD + n], nW1[(kg + 1) * DD + n]);
        }
        __syncthreads();
        Frag8 af;
        const int abase = rloc16 * 132 + 16 * s + (hiHalf ? 4 : 0);
        af.q[0] = lds_q(&Ab32[abase]);
        af.q[1] = lds_q(&Ab32[abase + 8]);
#pragma unroll
        for (int nt = 0; nt < 8; ++nt) {
            Frag8 bf;
            const int bbase = (nt * 16 + rloc16) * 20 + (hiHalf ? 4 : 0);
            bf.q[0] = lds_q(&pkt[bbase]);
            bf.q[1] = lds_q(&pkt[bbase + 8]);
            acc[nt] = __builtin_amdgcn_wmma_f32_16x16x32_f16(
                false, af.h, false, bf.h, (short)0, acc[nt], false, false);
        }
    }

    // ---- LN over 128 + SiLU, write back f16 to Abuf cols 0..127 --------
    {
        float sv[8][8];
        float rs[8], rq[8];
#pragma unroll
        for (int r = 0; r < 8; ++r) { rs[r] = 0.0f; rq[r] = 0.0f; }
#pragma unroll
        for (int nt = 0; nt < 8; ++nt) {
            int col = nt * 16 + rloc16;
            float bb = nb1s[col];
#pragma unroll
            for (int r = 0; r < 8; ++r) {
                float t = acc[nt][r] + bb;
                sv[nt][r] = t; rs[r] += t; rq[r] += t * t;
            }
        }
#pragma unroll
        for (int r = 0; r < 8; ++r) {
            rs[r] += __shfl_xor(rs[r], 1, 32);
            rs[r] += __shfl_xor(rs[r], 2, 32);
            rs[r] += __shfl_xor(rs[r], 4, 32);
            rs[r] += __shfl_xor(rs[r], 8, 32);
            rq[r] += __shfl_xor(rq[r], 1, 32);
            rq[r] += __shfl_xor(rq[r], 2, 32);
            rq[r] += __shfl_xor(rq[r], 4, 32);
            rq[r] += __shfl_xor(rq[r], 8, 32);
        }
        const float inv128 = 1.0f / 128.0f;
#pragma unroll
        for (int r = 0; r < 8; ++r) {
            float mean = rs[r] * inv128;
            float var = rq[r] * inv128 - mean * mean;
            float inv = rsqrtf(var + 1e-5f);
            int row2 = r + (hiHalf ? 8 : 0);
#pragma unroll
            for (int nt = 0; nt < 8; ++nt) {
                int col = nt * 16 + rloc16;
                float h = silu_f((sv[nt][r] - mean) * inv * ngs[col]);
                Abuf[wave][row2 * 264 + col] = (_Float16)h;
            }
        }
    }

    // ---- GEMM2: K=128 in 4 k-slices ------------------------------------
    v8f acc2[8] = {};
    for (int s = 0; s < 4; ++s) {
        __syncthreads();
        for (int idx = tid; idx < 128 * 16; idx += 128) {
            int n = idx >> 4, kpl = idx & 15;
            int kg = 32 * s + 2 * kpl;
            pkt[n * 20 + kpl] = pkh(nW2[kg * DD + n], nW2[(kg + 1) * DD + n]);
        }
        __syncthreads();
        Frag8 af;
        const int abase = rloc16 * 132 + 16 * s + (hiHalf ? 4 : 0);
        af.q[0] = lds_q(&Ab32[abase]);
        af.q[1] = lds_q(&Ab32[abase + 8]);
#pragma unroll
        for (int nt = 0; nt < 8; ++nt) {
            Frag8 bf;
            const int bbase = (nt * 16 + rloc16) * 20 + (hiHalf ? 4 : 0);
            bf.q[0] = lds_q(&pkt[bbase]);
            bf.q[1] = lds_q(&pkt[bbase + 8]);
            acc2[nt] = __builtin_amdgcn_wmma_f32_16x16x32_f16(
                false, af.h, false, bf.h, (short)0, acc2[nt], false, false);
        }
    }

    // ---- write output ---------------------------------------------------
#pragma unroll
    for (int nt = 0; nt < 8; ++nt) {
        int col = nt * 16 + rloc16;
        float bb = nb2s[col];
#pragma unroll
        for (int r = 0; r < 8; ++r) {
            int row = rowblock + wave * 16 + r + (hiHalf ? 8 : 0);
            out[(size_t)row * DD + col] = acc2[nt][r] + bb;
        }
    }
}

// ---------------------------------------------------------------------------
extern "C" void kernel_launch(void* const* d_in, const int* in_sizes, int n_in,
                              void* d_out, int out_size, void* d_ws, size_t ws_size,
                              hipStream_t stream) {
    const int*           atoms    = (const int*)d_in[0];
    const float*         rel      = (const float*)d_in[1];
    const unsigned char* adj      = (const unsigned char*)d_in[2];
    const unsigned char* mask     = (const unsigned char*)d_in[3];
    const float*         softmask = (const float*)d_in[4];
    const float*         atom_tab = (const float*)d_in[5];
    const float*         neigh_tab= (const float*)d_in[6];
    const float* rW1 = (const float*)d_in[7];
    const float* rb1 = (const float*)d_in[8];
    const float* rg1 = (const float*)d_in[9];
    const float* rW2 = (const float*)d_in[10];
    const float* rb2 = (const float*)d_in[11];
    const float* rg2 = (const float*)d_in[12];
    const float* rW3 = (const float*)d_in[13];
    const float* rb3 = (const float*)d_in[14];
    const float* nW1 = (const float*)d_in[15];
    const float* nb1 = (const float*)d_in[16];
    const float* ngv = (const float*)d_in[17];
    const float* nW2 = (const float*)d_in[18];
    const float* nb2 = (const float*)d_in[19];

    float* out = (float*)d_out;

    // workspace layout: nembB [B*N*D] f32, then nf [B*N*D] f32
    float* nembB = (float*)d_ws;
    float* nf    = nembB + (size_t)BB * NN * DD;

    nsf_prep_nemb<<<BB * NN, DD, 0, stream>>>(atoms, neigh_tab, nembB);

    nsf_pair_kernel<<<BB * NN, 256, 0, stream>>>(
        rel, adj, mask, softmask,
        rW1, rb1, rg1, rW2, rb2, rg2, rW3, rb3,
        nembB, nf);

    nsf_node_kernel<<<(BB * NN) / 64, 128, 0, stream>>>(
        atoms, atom_tab, nf, nW1, nb1, ngv, nW2, nb2, out);
}
